// Tokenizer_49400713838737
// MI455X (gfx1250) — compile-verified, matrix operand order checked by reading, not run on hardware
//
#include <hip/hip_runtime.h>

typedef _Float16 f16;
typedef __attribute__((ext_vector_type(8)))  _Float16 v8h;
typedef __attribute__((ext_vector_type(16))) _Float16 v16h;
typedef __attribute__((ext_vector_type(8)))  float    v8f;
typedef __attribute__((ext_vector_type(4)))  float    v4f;
typedef __attribute__((ext_vector_type(4)))  int      v4i;

#define B_    32
#define V_    8
#define S_    8256
#define LAGS_ 4
#define T_    512          // (S - W) / STEP
#define CO_   16           // CONV_OUT
#define E_    256
#define K_    512
#define BT_   (B_ * T_)    // 16384
#define CN_   (S_ / LAGS_) // 2064 chunks per batch row
#define KLIN_ (E_ * CO_)   // 4096

// ---------------------------------------------------------------------------
// Repack lin_w[j, ch*16 + p]  ->  lin_w2[j, p*256 + ch]  (f16).
// With this layout, the K-slab [p*256, (p+1)*256) of GEMM row t is the
// contiguous chunk-feature vector chunkfeat[b, t*4+p, :].
// ---------------------------------------------------------------------------
__global__ void prep_linw_kernel(const float* __restrict__ lin_w,
                                 f16* __restrict__ lin_w2) {
    int g  = blockIdx.x * 256 + threadIdx.x;   // j*4096 + p*256 + ch
    int j  = g >> 12;
    int kp = g & 4095;
    int p  = kp >> 8;
    int ch = kp & 255;
    lin_w2[g] = (f16)lin_w[(size_t)j * KLIN_ + ch * CO_ + p];
}

// Codebook -> f16 copy + per-code squared norm (|e|^2 is row-constant, drop it)
__global__ void prep_cb_kernel(const float* __restrict__ cb,
                               f16* __restrict__ cb16,
                               float* __restrict__ cbn) {
    __shared__ float red[256];
    int k = blockIdx.x, i = threadIdx.x;
    float v = cb[(size_t)k * E_ + i];
    cb16[(size_t)k * E_ + i] = (f16)v;
    red[i] = v * v;
    __syncthreads();
    for (int s = 128; s > 0; s >>= 1) {
        if (i < s) red[i] += red[i + s];
        __syncthreads();
    }
    if (i == 0) cbn[k] = red[0];
}

// ---------------------------------------------------------------------------
// Stride-4 conv de-duplicated across overlapping windows: one block per
// (b, chunk); chunk c uses x[b, :, 4c..4c+3].  Output relu(conv+bias) as f16.
// ---------------------------------------------------------------------------
__global__ void conv_chunk_kernel(const float* __restrict__ x,
                                  const float* __restrict__ conv_w,
                                  const float* __restrict__ conv_b,
                                  f16* __restrict__ chunkfeat) {
    __shared__ float xs[V_ * LAGS_];
    int blk = blockIdx.x;
    int b = blk / CN_;
    int c = blk % CN_;
    int tid = threadIdx.x;
    if (tid < V_ * LAGS_) {
        int v = tid >> 2, l = tid & 3;
        xs[tid] = x[(size_t)b * V_ * S_ + (size_t)v * S_ + 4 * c + l];
    }
    __syncthreads();
    float acc = conv_b[tid];
    const float* wrow = conv_w + tid * (V_ * LAGS_);
#pragma unroll
    for (int i = 0; i < V_ * LAGS_; ++i) acc += xs[i] * wrow[i];
    acc = acc > 0.f ? acc : 0.f;
    chunkfeat[(size_t)blk * E_ + tid] = (f16)acc;
}

// ---------------------------------------------------------------------------
// e = relu_feats @ lin_w2^T + lin_b.  M=16384, N=256, K=4096.
// Block: 512 threads = 16 waves; wave w owns N-tile [w*16, w*16+16).
// Per p (0..15) stage the contiguous 16x256 f16 A panel in LDS, then 8
// v_wmma_f32_16x16x32_f16 steps.  CDNA5 16-bit fragment layout:
//   lane m=lane&15 (M/N), kgroup=lane>>4; halves [0..7]=K kg*8.., [8..15]=K 16+kg*8..
// ---------------------------------------------------------------------------
__global__ __launch_bounds__(512)
void gemm_e_kernel(const f16* __restrict__ chunkfeat,
                   const f16* __restrict__ lin_w2,
                   const float* __restrict__ lin_b,
                   f16* __restrict__ e16) {
    __shared__ __align__(16) f16 As[16][E_];  // 8 KB
    const int tid  = threadIdx.x;
    const int wv   = tid >> 5;
    const int lane = tid & 31;
    const int bt0  = blockIdx.x * 16;
    const int b    = bt0 / T_;
    const int t0   = bt0 % T_;       // 16-row tile stays inside one batch row
    const int m    = lane & 15;
    const int kg   = lane >> 4;
    const int j    = wv * 16 + m;    // output channel (N index)
    const int col8 = lane * 8;

    v8f acc = {0.f, 0.f, 0.f, 0.f, 0.f, 0.f, 0.f, 0.f};

    for (int p = 0; p < CO_; ++p) {
        __syncthreads();
        // stage A panel: row wv (=tid>>5), 16 bytes per thread, coalesced
        {
            const f16* src =
                chunkfeat + ((size_t)(b * CN_ + (t0 + wv) * 4 + p)) * E_ + col8;
            *(v4i*)&As[wv][col8] = *(const v4i*)src;
        }
        __syncthreads();
#pragma unroll
        for (int kk = 0; kk < 8; ++kk) {
            const int kb = kk * 32;
            v8h a0 = *(const v8h*)&As[m][kb + kg * 8];
            v8h a1 = *(const v8h*)&As[m][kb + 16 + kg * 8];
            v16h a = __builtin_shufflevector(a0, a1, 0, 1, 2, 3, 4, 5, 6, 7, 8,
                                             9, 10, 11, 12, 13, 14, 15);
            const f16* bp = lin_w2 + (size_t)j * KLIN_ + p * E_ + kb + kg * 8;
            v8h b0 = *(const v8h*)bp;
            v8h b1 = *(const v8h*)(bp + 16);
            v16h bb = __builtin_shufflevector(b0, b1, 0, 1, 2, 3, 4, 5, 6, 7, 8,
                                              9, 10, 11, 12, 13, 14, 15);
            acc = __builtin_amdgcn_wmma_f32_16x16x32_f16(
                false, a, false, bb, (short)0, acc, false, false);
        }
    }

    const float bias = lin_b[j];
#pragma unroll
    for (int r = 0; r < 8; ++r) {
        int row = r + 8 * kg;  // C layout: vgpr r -> M=r (lanes 0-15) / 8+r
        e16[(size_t)(bt0 + row) * E_ + j] = (f16)(acc[r] + bias);
    }
}

// ---------------------------------------------------------------------------
// score[k] = |cb_k|^2 - 2 e.cb_k  (|e|^2 dropped: row-constant), argmin over
// 512 codes, out row = codebook[idx] (exact f32 copy).
// Block: 16 rows x 512 codes; wave w owns code tiles w*32 and w*32+16.
// ---------------------------------------------------------------------------
__global__ __launch_bounds__(512)
void dist_argmin_kernel(const f16* __restrict__ e16,
                        const f16* __restrict__ cb16,
                        const float* __restrict__ cbn,
                        const float* __restrict__ codebook,
                        float* __restrict__ out) {
    __shared__ float sc[16][K_];  // 32 KB
    const int tid  = threadIdx.x;
    const int wv   = tid >> 5;
    const int lane = tid & 31;
    const int bt0  = blockIdx.x * 16;
    const int m    = lane & 15;
    const int kg   = lane >> 4;
    const int j0   = wv * 32 + m;
    const int j1   = j0 + 16;

    v8f acc0 = {0.f, 0.f, 0.f, 0.f, 0.f, 0.f, 0.f, 0.f};
    v8f acc1 = {0.f, 0.f, 0.f, 0.f, 0.f, 0.f, 0.f, 0.f};

#pragma unroll
    for (int kk = 0; kk < 8; ++kk) {
        const int kb = kk * 32;
        const f16* ap = e16 + (size_t)(bt0 + m) * E_ + kb + kg * 8;
        v8h a0 = *(const v8h*)ap;
        v8h a1 = *(const v8h*)(ap + 16);
        v16h a = __builtin_shufflevector(a0, a1, 0, 1, 2, 3, 4, 5, 6, 7, 8, 9,
                                         10, 11, 12, 13, 14, 15);
        const f16* bp0 = cb16 + (size_t)j0 * E_ + kb + kg * 8;
        v8h c0 = *(const v8h*)bp0;
        v8h c1 = *(const v8h*)(bp0 + 16);
        v16h bb0 = __builtin_shufflevector(c0, c1, 0, 1, 2, 3, 4, 5, 6, 7, 8, 9,
                                           10, 11, 12, 13, 14, 15);
        acc0 = __builtin_amdgcn_wmma_f32_16x16x32_f16(
            false, a, false, bb0, (short)0, acc0, false, false);
        const f16* bp1 = cb16 + (size_t)j1 * E_ + kb + kg * 8;
        v8h d0 = *(const v8h*)bp1;
        v8h d1 = *(const v8h*)(bp1 + 16);
        v16h bb1 = __builtin_shufflevector(d0, d1, 0, 1, 2, 3, 4, 5, 6, 7, 8, 9,
                                           10, 11, 12, 13, 14, 15);
        acc1 = __builtin_amdgcn_wmma_f32_16x16x32_f16(
            false, a, false, bb1, (short)0, acc1, false, false);
    }

    const float cn0 = cbn[j0];
    const float cn1 = cbn[j1];
#pragma unroll
    for (int r = 0; r < 8; ++r) {
        int row = r + 8 * kg;
        sc[row][j0] = cn0 - 2.f * acc0[r];
        sc[row][j1] = cn1 - 2.f * acc1[r];
    }
    __syncthreads();

    // wave wv reduces row wv: each lane scans 16 strided columns, then a
    // 5-step xor-shuffle min with (value, lowest-index) tie-break.
    const int row = wv;
    float bv = 3.4e38f;
    int   bi = 0;
#pragma unroll
    for (int i = 0; i < K_ / 32; ++i) {
        int c = lane + i * 32;
        float v = sc[row][c];
        if (v < bv) { bv = v; bi = c; }
    }
#pragma unroll
    for (int off = 16; off >= 1; off >>= 1) {
        float ov = __shfl_xor(bv, off, 32);
        int   oi = __shfl_xor(bi, off, 32);
        if (ov < bv || (ov == bv && oi < bi)) { bv = ov; bi = oi; }
    }
    // all lanes hold the winning index: gather the f32 codebook row
    const float* src = codebook + (size_t)bi * E_ + lane * 8;
    float* dst = out + (size_t)(bt0 + row) * E_ + lane * 8;
    *(v4f*)dst       = *(const v4f*)src;
    *(v4f*)(dst + 4) = *(const v4f*)(src + 4);
}

extern "C" void kernel_launch(void* const* d_in, const int* in_sizes, int n_in,
                              void* d_out, int out_size, void* d_ws,
                              size_t ws_size, hipStream_t stream) {
    (void)in_sizes; (void)n_in; (void)out_size; (void)ws_size;
    const float* x        = (const float*)d_in[0];
    const float* conv_w   = (const float*)d_in[1];
    const float* conv_b   = (const float*)d_in[2];
    const float* lin_w    = (const float*)d_in[3];
    const float* lin_b    = (const float*)d_in[4];
    const float* codebook = (const float*)d_in[5];
    float* out = (float*)d_out;

    // workspace layout (all 16B-aligned; total ~44.6 MB)
    char* ws = (char*)d_ws;
    f16* chunkfeat = (f16*)ws;
    size_t off = (size_t)B_ * CN_ * E_ * sizeof(f16);       // 33.8 MB
    f16* lin_w2 = (f16*)(ws + off);
    off += (size_t)E_ * KLIN_ * sizeof(f16);                //  2.0 MB
    f16* e16 = (f16*)(ws + off);
    off += (size_t)BT_ * E_ * sizeof(f16);                  //  8.4 MB
    f16* cb16 = (f16*)(ws + off);
    off += (size_t)K_ * E_ * sizeof(f16);                   //  0.26 MB
    float* cbn = (float*)(ws + off);                        //  2 KB

    prep_linw_kernel<<<(E_ * KLIN_) / 256, 256, 0, stream>>>(lin_w, lin_w2);
    prep_cb_kernel<<<K_, 256, 0, stream>>>(codebook, cb16, cbn);
    conv_chunk_kernel<<<B_ * CN_, 256, 0, stream>>>(x, conv_w, conv_b,
                                                    chunkfeat);
    gemm_e_kernel<<<BT_ / 16, 512, 0, stream>>>(chunkfeat, lin_w2, lin_b, e16);
    dist_argmin_kernel<<<BT_ / 16, 512, 0, stream>>>(e16, cb16, cbn, codebook,
                                                     out);
}
